// SemiSinkhornKnopp_87814901334083
// MI455X (gfx1250) — matrix operand order repaired and checked by reading, not compiled, not run on hardware
//
#include <hip/hip_runtime.h>

typedef __bf16 bf16_t;
typedef __attribute__((ext_vector_type(16))) __bf16 v16bf;
typedef __attribute__((ext_vector_type(8)))  __bf16 v8bf;
typedef __attribute__((ext_vector_type(4)))  __bf16 v4bf;
typedef __attribute__((ext_vector_type(8)))  float  v8f;
typedef __attribute__((ext_vector_type(4)))  float  v4f;

namespace {
constexpr int    NROWS    = 65536;
constexpr int    KCOLS    = 1024;
constexpr float  INV_EPS  = 10.0f;            // 1/eps, eps = 0.1
constexpr float  FI       = 1.0f / 1.1f;      // gamma/(gamma+eps)
constexpr float  PA       = 1.0f / 65536.0f;
constexpr float  PB       = 1.0f / 1024.0f;
constexpr float  STOPERR  = 1e-6f;
constexpr int    MAX_ITER = 1000;

constexpr int    ATQ_BLOCKS = 1024;
constexpr int    ATQ_ROWS   = NROWS / ATQ_BLOCKS;   // 64 rows per block

// workspace layout (bytes)
constexpr size_t OFF_Q    = 0;                                          // bf16 Q, 134 MB (fits 192 MB L2)
constexpr size_t OFF_A    = OFF_Q    + (size_t)NROWS * KCOLS * 2;       // fp32 a[N]
constexpr size_t OFF_B    = OFF_A    + (size_t)NROWS * 4;               // fp32 b[K]
constexpr size_t OFF_BBF  = OFF_B    + (size_t)KCOLS * 4;               // bf16 b[K]
constexpr size_t OFF_ACC  = OFF_BBF  + (size_t)KCOLS * 2;               // fp32 acc[K]   (Q^T a)
constexpr size_t OFF_WRAW = OFF_ACC  + (size_t)KCOLS * 4;               // fp32 wraw[K]
constexpr size_t OFF_SCAL = OFF_WRAW + (size_t)KCOLS * 4;               // flag, counter, loss
} // namespace

// ---------------- init: reset all iteration state (deterministic across replays) ---
__global__ void init_kernel(float* __restrict__ b, bf16_t* __restrict__ bbf,
                            float* __restrict__ acc, float* __restrict__ wraw,
                            int* __restrict__ flag, int* __restrict__ counter,
                            float* __restrict__ loss) {
    int k = threadIdx.x;               // 1024 threads
    b[k]    = PB;
    bbf[k]  = (bf16_t)PB;
    acc[k]  = 0.0f;
    wraw[k] = PB;
    if (k == 0) { *flag = 0; *counter = 0; *loss = 0.0f; }
}

// ---------------- build Q in bf16: Q = max(exp(-P/eps), 1e-12) ---------------------
__global__ __launch_bounds__(256)
void build_q_kernel(const float* __restrict__ P, bf16_t* __restrict__ Q) {
    size_t base = ((size_t)blockIdx.x * 256 + threadIdx.x) * 8;
    v4f p0 = *(const v4f*)(P + base);
    v4f p1 = *(const v4f*)(P + base + 4);
    v8bf q;
    q[0] = (bf16_t)fmaxf(__expf(-p0[0] * INV_EPS), 1e-12f);
    q[1] = (bf16_t)fmaxf(__expf(-p0[1] * INV_EPS), 1e-12f);
    q[2] = (bf16_t)fmaxf(__expf(-p0[2] * INV_EPS), 1e-12f);
    q[3] = (bf16_t)fmaxf(__expf(-p0[3] * INV_EPS), 1e-12f);
    q[4] = (bf16_t)fmaxf(__expf(-p1[0] * INV_EPS), 1e-12f);
    q[5] = (bf16_t)fmaxf(__expf(-p1[1] * INV_EPS), 1e-12f);
    q[6] = (bf16_t)fmaxf(__expf(-p1[2] * INV_EPS), 1e-12f);
    q[7] = (bf16_t)fmaxf(__expf(-p1[3] * INV_EPS), 1e-12f);
    *(v8bf*)(Q + base) = q;
}

// ---------------- pass 1: a = Pa / max(Q @ b, 1e-12) via WMMA bf16 -----------------
// One wave32 per 16-row tile. b broadcast into all 16 B-columns, so every output
// column of the 16x16 D tile equals the GEMV result for its row.
__global__ __launch_bounds__(128)
void gemv_qb_kernel(const bf16_t* __restrict__ Q, const bf16_t* __restrict__ bbf,
                    float* __restrict__ a, const int* __restrict__ flag) {
    if (*flag) return;                               // converged: no-op (uniform)
    int lane  = threadIdx.x & 31;
    int wave  = threadIdx.x >> 5;
    int tile  = blockIdx.x * 4 + wave;               // 4096 tiles total
    int row0  = tile * 16;
    int m     = lane & 15;
    int half  = lane >> 4;                           // 0: lanes 0-15, 1: lanes 16-31
    int kA    = half * 8;                            // A: K sub-block offset per lane
    int kB    = half * 16;                           // unused; B loads use kA, same k order as A
    const bf16_t* qrow = Q + (size_t)(row0 + m) * KCOLS;

    v8f c = {};
    for (int k0 = 0; k0 < KCOLS; k0 += 32) {
        union { v16bf v; v8bf h[2]; } A, B;
        A.h[0] = *(const v8bf*)(qrow + k0 + kA);          // K = k0+kA .. +7
        A.h[1] = *(const v8bf*)(qrow + k0 + 16 + kA);     // K = k0+16+kA .. +7
        B.h[0] = *(const v8bf*)(bbf + k0 + kA);           // broadcast b into all cols
        B.h[1] = *(const v8bf*)(bbf + k0 + 16 + kA);
        c = __builtin_amdgcn_wmma_f32_16x16x32_bf16(
                false, A.v, false, B.v, (short)0, c, false, false);
    }
    // D layout: lanes 0-15 hold rows 0-7 (c[r] = row r), lanes 16-31 hold rows 8-15.
    // All columns identical, so lane 0 / lane 16 each write 8 rows.
    if ((lane & 15) == 0) {
        int rbase = row0 + half * 8;
#pragma unroll
        for (int r = 0; r < 8; ++r)
            a[rbase + r] = PA / fmaxf(c[r], 1e-12f);
    }
}

// ---------------- pass 2 (fused): acc = Q^T a, then last block updates b -----------
__global__ __launch_bounds__(256)
void gemv_qta_kernel(const bf16_t* __restrict__ Q, const float* __restrict__ a,
                     float* __restrict__ acc, float* __restrict__ b,
                     bf16_t* __restrict__ bbf, float* __restrict__ wraw,
                     int* __restrict__ flag, int* __restrict__ counter) {
    if (*flag) return;
    int tid   = threadIdx.x;
    int kbase = tid * 4;                             // this thread owns cols kbase..+3
    int row0  = blockIdx.x * ATQ_ROWS;
    float r0 = 0.f, r1 = 0.f, r2 = 0.f, r3 = 0.f;
    for (int r = 0; r < ATQ_ROWS; ++r) {
        int row  = row0 + r;
        float av = a[row];
        v4bf q = *(const v4bf*)(Q + (size_t)row * KCOLS + kbase);
        r0 += av * (float)q[0];
        r1 += av * (float)q[1];
        r2 += av * (float)q[2];
        r3 += av * (float)q[3];
    }
    atomicAdd(&acc[kbase + 0], r0);
    atomicAdd(&acc[kbase + 1], r1);
    atomicAdd(&acc[kbase + 2], r2);
    atomicAdd(&acc[kbase + 3], r3);

    __threadfence();
    __shared__ int isLast;
    if (tid == 0) isLast = (atomicAdd(counter, 1) == (int)gridDim.x - 1);
    __syncthreads();
    if (!isLast) return;
    __threadfence();

    // Finisher block: b_new = (Pb / max(Q^T a,1e-12))^fi, err check, w_raw, re-zero.
    __shared__ float red[256];
    float errsq = 0.f;
#pragma unroll
    for (int j = 0; j < 4; ++j) {
        int k = kbase + j;
        float s = __hip_atomic_load(&acc[k], __ATOMIC_RELAXED, __HIP_MEMORY_SCOPE_AGENT);
        float bn = __powf(PB / fmaxf(s, 1e-12f), FI);
        float d  = bn - b[k];
        errsq   += d * d;
        b[k]     = bn;
        bbf[k]   = (bf16_t)bn;
        wraw[k]  = bn * s;        // = N * mean(OT_plan, axis=0)[k], computed for free
        acc[k]   = 0.0f;          // ready for next iteration
    }
    red[tid] = errsq;
    __syncthreads();
    for (int off = 128; off > 0; off >>= 1) {
        if (tid < off) red[tid] += red[tid + off];
        __syncthreads();
    }
    if (tid == 0) {
        *counter = 0;
        if (red[0] <= STOPERR * STOPERR) *flag = 1;
    }
}

// ---------------- finalize: OT_plan (fp32, exact Q) + ot_loss partials -------------
__global__ __launch_bounds__(256)
void finalize_kernel(const float* __restrict__ P, const float* __restrict__ a,
                     const float* __restrict__ b, float* __restrict__ OT,
                     float* __restrict__ loss) {
    int row = blockIdx.x;                            // one row per block
    const float* prow = P + (size_t)row * KCOLS;
    float*       orow = OT + (size_t)row * KCOLS;
    float av = (float)NROWS * a[row];
    float part = 0.f;
    for (int k = threadIdx.x; k < KCOLS; k += 256) {
        float p  = prow[k];
        float q  = fmaxf(__expf(-p * INV_EPS), 1e-12f);
        float ot = av * q * b[k];
        orow[k]  = ot;
        part    += ot * p;
    }
    __shared__ float red[256];
    red[threadIdx.x] = part;
    __syncthreads();
    for (int off = 128; off > 0; off >>= 1) {
        if (threadIdx.x < off) red[threadIdx.x] += red[threadIdx.x + off];
        __syncthreads();
    }
    if (threadIdx.x == 0) atomicAdd(loss, red[0]);
}

// ---------------- epilogue: ot_loss scalar + KL regularizer -----------------------
__global__ void epilogue_kernel(const float* __restrict__ wraw,
                                const float* __restrict__ loss,
                                float* __restrict__ out2) {
    __shared__ float red[1024];
    int k = threadIdx.x;
    float wm = wraw[k] * (1.0f / (float)NROWS);      // w = mean(OT_plan, axis=0)
    red[k] = wm;
    __syncthreads();
    for (int off = 512; off > 0; off >>= 1) {
        if (k < off) red[k] += red[k + off];
        __syncthreads();
    }
    float denom = red[0] + 1e-8f;
    __syncthreads();
    float wfin = wm / denom;
    red[k] = PB * (__logf(PB) - __logf(wfin + 1e-7f));
    __syncthreads();
    for (int off = 512; off > 0; off >>= 1) {
        if (k < off) red[k] += red[k + off];
        __syncthreads();
    }
    if (k == 0) {
        out2[0] = *loss * (1.0f / (float)NROWS);     // ot_loss
        out2[1] = red[0];                            // reg
    }
}

extern "C" void kernel_launch(void* const* d_in, const int* in_sizes, int n_in,
                              void* d_out, int out_size, void* d_ws, size_t ws_size,
                              hipStream_t stream) {
    const float* P   = (const float*)d_in[0];
    float*       out = (float*)d_out;
    char*        ws  = (char*)d_ws;

    bf16_t* Q       = (bf16_t*)(ws + OFF_Q);
    float*  a       = (float*)(ws + OFF_A);
    float*  b       = (float*)(ws + OFF_B);
    bf16_t* bbf     = (bf16_t*)(ws + OFF_BBF);
    float*  acc     = (float*)(ws + OFF_ACC);
    float*  wraw    = (float*)(ws + OFF_WRAW);
    int*    flag    = (int*)(ws + OFF_SCAL);
    int*    counter = flag + 1;
    float*  loss    = (float*)(flag + 2);

    init_kernel<<<1, 1024, 0, stream>>>(b, bbf, acc, wraw, flag, counter, loss);
    build_q_kernel<<<(NROWS * (size_t)KCOLS) / (256 * 8), 256, 0, stream>>>(P, Q);

    for (int it = 0; it < MAX_ITER; ++it) {
        gemv_qb_kernel<<<NROWS / 16 / 4, 128, 0, stream>>>(Q, bbf, a, flag);
        gemv_qta_kernel<<<ATQ_BLOCKS, 256, 0, stream>>>(Q, a, acc, b, bbf, wraw,
                                                        flag, counter);
    }

    finalize_kernel<<<NROWS, 256, 0, stream>>>(P, a, b, out, loss);
    epilogue_kernel<<<1, 1024, 0, stream>>>(wraw, loss, out + (size_t)NROWS * KCOLS);
}
